// Matcher_57861799411981
// MI455X (gfx1250) — matrix operand order repaired and checked by reading, not compiled, not run on hardware
//
#include <hip/hip_runtime.h>

typedef __attribute__((ext_vector_type(16))) _Float16 v16h;
typedef __attribute__((ext_vector_type(8)))  float    v8f;

union AFrag { v16h v; _Float16 e[16]; };
union CFrag { v8f  v; float     e[8];  };

#define BNEPS 1e-5f

// ---------------------------------------------------------------------------
// Implicit-GEMM conv / GEMM kernel using V_WMMA_F32_16X16X32_F16.
//   out[cout, p] = sum_k wq[cout, k] * patch(k, p)     (f16 in, f32 acc)
//   patch(k,p): k -> (ci, r, s) [S_ = conv kernel width, templated],
//               p -> (i, j); reads x[ci, i+r, j+s]
//   epilogue: v = acc*scale[cout] + shift[cout]; optional relu
//   output index: (b*Cout+cout)*outChStride + (i*ostep+oy)*OW + (j*ostep+ox)
//   aBatchStride: per-image stride of wq (0 for shared conv weights)
// Block = 128 threads (4 wave32). Block tile = 64 (cout) x 64 (pixels);
// each wave owns a 32x32 quadrant = 2x2 WMMA tiles (4 wmma per K-chunk).
// ---------------------------------------------------------------------------
template <int S_>
__global__ __launch_bounds__(128)
void conv_wmma(const float* __restrict__ x, const _Float16* __restrict__ wq,
               const float* __restrict__ scale, const float* __restrict__ shift,
               float* __restrict__ out,
               int Cin, int Hin, int Win, int Cout, int Ho, int Wo,
               int OW, int oy, int ox, int ostep,
               int outChStride, int relu, long long aBatchStride)
{
    constexpr int RS = S_ * S_;
    __shared__ _Float16 As[64][34];    // [cout][k]   (17-dword rows: conflict-free)
    __shared__ _Float16 Bs[64][34];    // [pixel][k]  (transposed: frag = contiguous)
    const int b  = blockIdx.z;
    const int P  = Ho * Wo;
    const int K  = Cin * RS;
    const int ct = blockIdx.y * 64;   // cout tile base
    const int pt = blockIdx.x * 64;   // pixel tile base
    const float*    xb = x  + (size_t)b * Cin * Hin * Win;
    const _Float16* wb = wq + (size_t)b * aBatchStride;
    const int tid  = threadIdx.x;
    const int lane = tid & 31;
    const int wv   = tid >> 5;            // wave id 0..3
    const int wr   = (wv >> 1) * 32;      // wave cout offset in tile
    const int wc   = (wv & 1) * 32;       // wave pixel offset in tile

    // B-stage per-thread invariants: fixed column, k = bsk0 + 2t
    const int  bsc  = tid & 63;
    const int  bsk0 = tid >> 6;           // 0 or 1
    const int  bp   = pt + bsc;
    const bool pok  = bp < P;
    int bi = 0, bj = 0;
    if (pok) { bi = bp / Wo; bj = bp - bi * Wo; }   // once, outside K loop

    CFrag acc[2][2];
#pragma unroll
    for (int a = 0; a < 2; ++a)
#pragma unroll
        for (int bb = 0; bb < 2; ++bb)
#pragma unroll
            for (int r = 0; r < 8; ++r) acc[a][bb].e[r] = 0.0f;

    for (int k0 = 0; k0 < K; k0 += 32) {
        // Stage A tile (64 couts x 32 k), coalesced f16 reads
#pragma unroll
        for (int t = 0; t < 16; ++t) {
            int e = tid + t * 128;
            int r = e >> 5, k = e & 31;
            int gc = ct + r, gk = k0 + k;
            _Float16 v = (_Float16)0.0f;
            if (gc < Cout && gk < K) v = wb[(size_t)gc * K + gk];
            As[r][k] = v;
        }
        // Stage B tile (32 k x 64 pixels) transposed; divisions are by
        // compile-time constants (magic-multiply; none at all for S_=1)
#pragma unroll
        for (int t = 0; t < 16; ++t) {
            int k  = bsk0 + 2 * t;
            int gk = k0 + k;
            _Float16 v = (_Float16)0.0f;
            if (pok && gk < K) {
                int ci, rr, ss;
                if constexpr (RS == 1) { ci = gk; rr = 0; ss = 0; }
                else { ci = gk / RS; int rs = gk - ci * RS; rr = rs / S_; ss = rs - rr * S_; }
                v = (_Float16)xb[((size_t)ci * Hin + (bi + rr)) * Win + (bj + ss)];
            }
            Bs[bsc][k] = v;
        }
        __syncthreads();

        // A fragment (contiguous k order, matches B):
        //   lane<16: row=lane,    K = 0..15
        //   lane>=16: row=lane-16, K = 16..31
        AFrag af[2], bf[2];
        const int kh = (lane >> 4) * 8;
        const int kb = (lane >> 4) * 16;
#pragma unroll
        for (int a = 0; a < 2; ++a) {
            const int arow = wr + a * 16 + (lane & 15);
#pragma unroll
            for (int jj = 0; jj < 16; ++jj) {
                af[a].e[jj]     = As[arow][kb + jj];
            }
        }
        // B fragment: col=lane&15 (+tile offset); lanes<16 hold K 0..15,
        // lanes>=16 hold K 16..31 -> 16 contiguous halves in Bs[col][]
#pragma unroll
        for (int bb = 0; bb < 2; ++bb) {
            const int bcol = wc + bb * 16 + (lane & 15);
#pragma unroll
            for (int jj = 0; jj < 16; ++jj) bf[bb].e[jj] = Bs[bcol][kb + jj];
        }
#pragma unroll
        for (int a = 0; a < 2; ++a)
#pragma unroll
            for (int bb = 0; bb < 2; ++bb)
                acc[a][bb].v = __builtin_amdgcn_wmma_f32_16x16x32_f16(
                    false, af[a].v, false, bf[bb].v, (short)0, acc[a][bb].v, false, false);

        // prefetch next A chunk (global_prefetch_b8) to cover weight latency
        if (k0 + 32 < K) {
            int gc = ct + bsc;
            if (gc < Cout)
                __builtin_prefetch((const void*)(wb + (size_t)gc * K + (k0 + 32)), 0, 1);
        }
        __syncthreads();
    }

    // Epilogue: C/D layout -> VGPR r holds M = 8*(lane>>4)+r, N = lane&15
#pragma unroll
    for (int bb = 0; bb < 2; ++bb) {
        const int N = lane & 15;
        const int p = pt + wc + bb * 16 + N;
        if (p >= P) continue;
        const int i = p / Wo, j = p - i * Wo;
        const size_t spat = (size_t)(i * ostep + oy) * OW + (j * ostep + ox);
#pragma unroll
        for (int a = 0; a < 2; ++a) {
#pragma unroll
            for (int r = 0; r < 8; ++r) {
                int M  = (lane >> 4) * 8 + r;
                int gc = ct + wr + a * 16 + M;
                if (gc < Cout) {
                    float v  = acc[a][bb].e[r];
                    float sc = scale ? scale[gc] : 1.0f;
                    float sh = shift ? shift[gc] : 0.0f;
                    v = v * sc + sh;
                    if (relu) v = fmaxf(v, 0.0f);
                    out[((size_t)b * Cout + gc) * (size_t)outChStride + spat] = v;
                }
            }
        }
    }
}

// --------------------------- small helper kernels ---------------------------

__global__ __launch_bounds__(256)
void cvt_f16(const float* __restrict__ in, _Float16* __restrict__ out, long long n)
{
    long long i = (long long)blockIdx.x * 256 + threadIdx.x;
    if (i < n) out[i] = (_Float16)in[i];
}

// ConvTranspose weight (C,O,2,2) -> 4 slices of [O x C] f16
__global__ __launch_bounds__(256)
void pack_wt(const float* __restrict__ w, _Float16* __restrict__ out, int C, int O)
{
    int idx = blockIdx.x * 256 + threadIdx.x;
    int total = 4 * O * C;
    if (idx >= total) return;
    int kl = idx / (O * C);
    int rem = idx - kl * (O * C);
    int o = rem / C, c = rem - o * C;
    int k = kl >> 1, l = kl & 1;
    out[idx] = (_Float16)w[((size_t)(c * O + o) * 2 + k) * 2 + l];
}

// activation [B][C][N] f32 -> A operand [B][N][C] f16 (per-image transpose)
__global__ __launch_bounds__(256)
void pack_at(const float* __restrict__ s, _Float16* __restrict__ a, int C, int N)
{
    int idx = blockIdx.x * 256 + threadIdx.x;
    if (idx >= C * N) return;
    int b = blockIdx.y;
    int n = idx / C, c = idx - n * C;
    a[((size_t)b * N + n) * C + c] = (_Float16)s[((size_t)b * C + c) * N + n];
}

__global__ __launch_bounds__(256)
void bn_prep(const float* __restrict__ g, const float* __restrict__ be,
             const float* __restrict__ m, const float* __restrict__ v,
             const float* __restrict__ bias, float* __restrict__ scale,
             float* __restrict__ shift, int C)
{
    int c = blockIdx.x * 256 + threadIdx.x;
    if (c >= C) return;
    float s = g[c] * rsqrtf(v[c] + BNEPS);
    scale[c] = s;
    shift[c] = be[c] - m[c] * s + bias[c] * s;
}

__global__ __launch_bounds__(256)
void maxpool2(const float* __restrict__ in, float* __restrict__ out,
              int C, int H, int W, int H2, int W2)
{
    long long idx = (long long)blockIdx.x * 256 + threadIdx.x;
    long long tot = (long long)C * H2 * W2;
    if (idx >= tot) return;
    int b = blockIdx.y;
    int c = (int)(idx / (H2 * W2));
    int r = (int)(idx - (long long)c * H2 * W2);
    int i = r / W2, j = r - i * W2;
    const float* p = in + ((size_t)b * C + c) * H * W + (size_t)(2 * i) * W + 2 * j;
    float v = fmaxf(fmaxf(p[0], p[1]), fmaxf(p[W], p[W + 1]));
    out[((size_t)b * C + c) * (size_t)(H2 * W2) + r] = v;
}

__global__ __launch_bounds__(256)
void tnorm(const float* __restrict__ t, float* __restrict__ out, int C, int M)
{
    int m = blockIdx.x * 256 + threadIdx.x;
    if (m >= M) return;
    int b = blockIdx.y;
    const float* tb = t + (size_t)b * C * M;
    float s = 0.0f;
    for (int c = 0; c < C; ++c) { float v = tb[(size_t)c * M + m]; s += v * v; }
    out[(size_t)b * M + m] = s;
}

// argmin_m (||t_m||^2 - 2 G[n,m]); write [src ; t[:,argmin]] into out channels
__global__ __launch_bounds__(256)
void nn_concat(const float* __restrict__ s, const float* __restrict__ t,
               const float* __restrict__ G, const float* __restrict__ tn2,
               float* __restrict__ out, int C, int N, int M, int outTotC, int outOff)
{
    int n = blockIdx.x * 256 + threadIdx.x;
    if (n >= N) return;
    int b = blockIdx.y;
    const float* Gr  = G + ((size_t)b * N + n) * M;
    const float* tnb = tn2 + (size_t)b * M;
    float best = 3.4e38f; int bi = 0;
    for (int m = 0; m < M; ++m) {
        float d = tnb[m] - 2.0f * Gr[m];
        if (d < best) { best = d; bi = m; }
    }
    const float* sb = s + (size_t)b * C * N;
    const float* tb = t + (size_t)b * C * M;
    float* ob = out + ((size_t)b * outTotC + outOff) * (size_t)N;
    for (int c = 0; c < C; ++c)       ob[(size_t)c * N + n]       = sb[(size_t)c * N + n];
    for (int c = 0; c < C; ++c)       ob[(size_t)(C + c) * N + n] = tb[(size_t)c * M + bi];
}

// half-pixel bilinear resize; writes into out at channel offset outOff of outTotC
__global__ __launch_bounds__(256)
void bilinear(const float* __restrict__ in, float* __restrict__ out,
              int C, int Hi, int Wi, int Ho, int Wo, int outTotC, int outOff)
{
    long long idx = (long long)blockIdx.x * 256 + threadIdx.x;
    long long tot = (long long)C * Ho * Wo;
    if (idx >= tot) return;
    int b = blockIdx.y;
    int c = (int)(idx / (Ho * Wo));
    int r = (int)(idx - (long long)c * Ho * Wo);
    int i = r / Wo, j = r - i * Wo;
    float fy = (i + 0.5f) * (float)Hi / (float)Ho - 0.5f;
    float fx = (j + 0.5f) * (float)Wi / (float)Wo - 0.5f;
    fy = fminf(fmaxf(fy, 0.0f), (float)(Hi - 1));
    fx = fminf(fmaxf(fx, 0.0f), (float)(Wi - 1));
    int y0 = (int)fy, x0 = (int)fx;
    int y1 = min(y0 + 1, Hi - 1), x1 = min(x0 + 1, Wi - 1);
    float wy = fy - y0, wx = fx - x0;
    const float* ib = in + ((size_t)b * C + c) * (size_t)(Hi * Wi);
    float v = (1.0f - wy) * ((1.0f - wx) * ib[(size_t)y0 * Wi + x0] + wx * ib[(size_t)y0 * Wi + x1])
            +         wy  * ((1.0f - wx) * ib[(size_t)y1 * Wi + x0] + wx * ib[(size_t)y1 * Wi + x1]);
    out[((size_t)b * outTotC + outOff + c) * (size_t)(Ho * Wo) + r] = v;
}

__global__ __launch_bounds__(256)
void conv1x1(const float* __restrict__ x, const float* __restrict__ w,
             const float* __restrict__ b0, float* __restrict__ out, int C, int P)
{
    int p = blockIdx.x * 256 + threadIdx.x;
    if (p >= P) return;
    int b = blockIdx.y;
    const float* xb = x + (size_t)b * C * P;
    float s = b0[0];
    for (int c = 0; c < C; ++c) s += xb[(size_t)c * P + p] * w[c];
    out[(size_t)b * P + p] = s;
}

// ---------------------------------------------------------------------------
extern "C" void kernel_launch(void* const* d_in, const int* in_sizes, int n_in,
                              void* d_out, int out_size, void* d_ws, size_t ws_size,
                              hipStream_t stream)
{
    (void)out_size; (void)ws_size;
    const int B = 4;

    // ---- parameter pointer resolution (handles 2 plausible flattenings) ----
    // logical enc idx: 0=W1 1=B1 2=W2 3=B2 4=G1 5=BE1 6=M1 7=V1 8=G2 9=BE2 10=M2 11=V2
    // logical dec idx: 0=W1 1=B1 2=W2 3=B2 4=WT 5=BT 6=G1 7=BE1 8=M1 9=V1 10=G2 11=BE2 12=M2 13=V2
    const float* enc[4][12];
    const float* dcp[2][14];     // dcp[0]=dec3, dcp[1]=dec2
    const float *dec1w, *dec1b, *src, *tar;
    int pbase, isrc, itar;
    if (in_sizes[0] == 262144 && in_sizes[1] == 262144) { isrc = 0; itar = 1; pbase = 2; }
    else { isrc = n_in - 2; itar = n_in - 1; pbase = 0; }
    src = (const float*)d_in[isrc];
    tar = (const float*)d_in[itar];
    auto gp = [&](int i) { return (const float*)d_in[pbase + i]; };
    bool sortedKeys = (in_sizes[pbase] == 1);   // dec1_b comes first when keys sorted
    if (sortedKeys) {
        dec1b = gp(0); dec1w = gp(1);
        int dbase[2] = {16, 2};                 // sorted: dec2 @2, dec3 @16
        for (int d = 0; d < 2; ++d) {
            int o = dbase[d];
            dcp[d][1]  = gp(o + 0);  dcp[d][3]  = gp(o + 1);  dcp[d][7]  = gp(o + 2);
            dcp[d][11] = gp(o + 3);  dcp[d][5]  = gp(o + 4);  dcp[d][6]  = gp(o + 5);
            dcp[d][10] = gp(o + 6);  dcp[d][8]  = gp(o + 7);  dcp[d][12] = gp(o + 8);
            dcp[d][9]  = gp(o + 9);  dcp[d][13] = gp(o + 10); dcp[d][0]  = gp(o + 11);
            dcp[d][2]  = gp(o + 12); dcp[d][4]  = gp(o + 13);
        }
        for (int i = 0; i < 4; ++i) {
            int o = 30 + 12 * i;
            enc[i][1]  = gp(o + 0); enc[i][3]  = gp(o + 1); enc[i][5] = gp(o + 2);
            enc[i][9]  = gp(o + 3); enc[i][4]  = gp(o + 4); enc[i][8] = gp(o + 5);
            enc[i][6]  = gp(o + 6); enc[i][10] = gp(o + 7); enc[i][7] = gp(o + 8);
            enc[i][11] = gp(o + 9); enc[i][0]  = gp(o + 10); enc[i][2] = gp(o + 11);
        }
    } else {                                    // insertion order
        for (int i = 0; i < 4; ++i)
            for (int k = 0; k < 12; ++k) enc[i][k] = gp(12 * i + k);
        for (int d = 0; d < 2; ++d)
            for (int k = 0; k < 14; ++k) dcp[d][k] = gp(48 + 14 * d + k);
        dec1w = gp(76); dec1b = gp(77);
    }

    // ------------------------------ workspace ------------------------------
    char* base = (char*)d_ws;
    size_t off = 0;
    auto allocF = [&](size_t n) { float* p = (float*)(base + off);
                                  off += ((n * 4 + 255) / 256) * 256; return p; };
    auto allocH = [&](size_t n) { _Float16* p = (_Float16*)(base + off);
                                  off += ((n * 2 + 255) / 256) * 256; return p; };

    float* X   = allocF((size_t)B * 64 * 254 * 254);   // big ping
    float* Y   = allocF((size_t)B * 64 * 252 * 252);   // big pong
    float* P1  = allocF((size_t)B * 64 * 126 * 126);   // pooled / d3t reuse
    float* P2  = allocF((size_t)B * 128 * 61 * 61);    // pooled / d1 reuse
    float* s3s = allocF((size_t)B * 256 * 784), *s3t = allocF((size_t)B * 256 * 784);
    float* s4s = allocF((size_t)B * 512 * 144), *s4t = allocF((size_t)B * 512 * 144);
    float* tn3 = allocF((size_t)B * 784),       *tn4 = allocF((size_t)B * 144);
    float* G3  = allocF((size_t)B * 784 * 784), *G4  = allocF((size_t)B * 144 * 144);
    float* dcat = allocF((size_t)B * 1536 * 784);
    float* c4   = allocF((size_t)B * 1024 * 144);
    _Float16* a3 = allocH((size_t)B * 784 * 256);
    _Float16* a4 = allocH((size_t)B * 144 * 512);

    const long long ew1n[4] = {576, 73728, 294912, 1179648};
    const long long ew2n[4] = {36864, 147456, 589824, 2359296};
    const long long dw1n[2] = {7077888, 294912};
    const long long dw2n[2] = {2359296, 147456};
    const long long dwtn[2] = {524288, 32768};
    _Float16 *ew1[4], *ew2[4], *dw1[2], *dw2[2], *dwt[2];
    for (int i = 0; i < 4; ++i) { ew1[i] = allocH(ew1n[i]); ew2[i] = allocH(ew2n[i]); }
    for (int d = 0; d < 2; ++d) { dw1[d] = allocH(dw1n[d]); dw2[d] = allocH(dw2n[d]);
                                  dwt[d] = allocH(dwtn[d]); }
    const int eC[4] = {64, 128, 256, 512};
    const int dC[2] = {512, 128};
    float *esc1[4], *esh1[4], *esc2[4], *esh2[4], *dsc1[2], *dsh1[2], *dsc2[2], *dsh2[2];
    for (int i = 0; i < 4; ++i) { esc1[i] = allocF(eC[i]); esh1[i] = allocF(eC[i]);
                                  esc2[i] = allocF(eC[i]); esh2[i] = allocF(eC[i]); }
    for (int d = 0; d < 2; ++d) { dsc1[d] = allocF(dC[d]); dsh1[d] = allocF(dC[d]);
                                  dsc2[d] = allocF(dC[d]); dsh2[d] = allocF(dC[d]); }

    // ------------------------------ launchers ------------------------------
    auto conv = [&](const float* x, const _Float16* wq, const float* sc, const float* sh,
                    float* out, int Cin, int Hin, int Win, int Cout, int Ho, int Wo,
                    int S, int OW, int oy, int ox, int ostep,
                    int outChStride, int relu, long long aStride) {
        dim3 g((Ho * Wo + 63) / 64, (Cout + 63) / 64, B);
        if (S == 3)
            conv_wmma<3><<<g, 128, 0, stream>>>(x, wq, sc, sh, out, Cin, Hin, Win,
                Cout, Ho, Wo, OW, oy, ox, ostep, outChStride, relu, aStride);
        else
            conv_wmma<1><<<g, 128, 0, stream>>>(x, wq, sc, sh, out, Cin, Hin, Win,
                Cout, Ho, Wo, OW, oy, ox, ostep, outChStride, relu, aStride);
    };
    auto cvt = [&](const float* w, _Float16* o, long long n) {
        cvt_f16<<<(unsigned)((n + 255) / 256), 256, 0, stream>>>(w, o, n);
    };
    auto bn = [&](const float* g, const float* be, const float* m, const float* v,
                  const float* bias, float* sc, float* sh, int C) {
        bn_prep<<<(C + 255) / 256, 256, 0, stream>>>(g, be, m, v, bias, sc, sh, C);
    };

    // -------------------------- weight / BN packing ------------------------
    for (int i = 0; i < 4; ++i) {
        cvt(enc[i][0], ew1[i], ew1n[i]);
        cvt(enc[i][2], ew2[i], ew2n[i]);
        bn(enc[i][4], enc[i][5], enc[i][6], enc[i][7], enc[i][1], esc1[i], esh1[i], eC[i]);
        bn(enc[i][8], enc[i][9], enc[i][10], enc[i][11], enc[i][3], esc2[i], esh2[i], eC[i]);
    }
    for (int d = 0; d < 2; ++d) {
        cvt(dcp[d][0], dw1[d], dw1n[d]);
        cvt(dcp[d][2], dw2[d], dw2n[d]);
        bn(dcp[d][6], dcp[d][7], dcp[d][8], dcp[d][9], dcp[d][1], dsc1[d], dsh1[d], dC[d]);
        bn(dcp[d][10], dcp[d][11], dcp[d][12], dcp[d][13], dcp[d][3], dsc2[d], dsh2[d], dC[d]);
    }
    pack_wt<<<(524288 + 255) / 256, 256, 0, stream>>>(dcp[0][4], dwt[0], 512, 256);
    pack_wt<<<(32768 + 255) / 256, 256, 0, stream>>>(dcp[1][4], dwt[1], 128, 64);

    // ------------------------------ encoders -------------------------------
    const int eH[4]  = {256, 126, 61, 28};
    const int eCi[4] = {1, 64, 128, 256};
    for (int im = 0; im < 2; ++im) {
        const float* cur = (im == 0) ? src : tar;
        for (int st = 0; st < 4; ++st) {
            int Hin = eH[st], H1 = Hin - 2, H2 = Hin - 4, Hp = H2 / 2;
            int Ci = eCi[st], Co = eC[st];
            conv(cur, ew1[st], esc1[st], esh1[st], X, Ci, Hin, Hin, Co, H1, H1,
                 3, H1, 0, 0, 1, H1 * H1, 1, 0);
            conv(X, ew2[st], esc2[st], esh2[st], Y, Co, H1, H1, Co, H2, H2,
                 3, H2, 0, 0, 1, H2 * H2, 1, 0);
            float* pooled = (st == 2) ? (im ? s3t : s3s)
                          : (st == 3) ? (im ? s4t : s4s)
                          : ((st == 0) ? P1 : P2);
            maxpool2<<<dim3((unsigned)((Co * Hp * Hp + 255) / 256), B), 256, 0, stream>>>(
                Y, pooled, Co, H2, H2, Hp, Hp);
            cur = pooled;
        }
    }

    // -------------------------- NN feature matching ------------------------
    // level 3: C=256, N=M=784
    pack_at<<<dim3((256 * 784 + 255) / 256, B), 256, 0, stream>>>(s3s, a3, 256, 784);
    tnorm<<<dim3((784 + 255) / 256, B), 256, 0, stream>>>(s3t, tn3, 256, 784);
    conv(s3t, a3, nullptr, nullptr, G3, 256, 1, 784, 784, 1, 784,
         1, 784, 0, 0, 1, 784, 0, (long long)784 * 256);
    nn_concat<<<dim3((784 + 255) / 256, B), 256, 0, stream>>>(
        s3s, s3t, G3, tn3, dcat, 256, 784, 784, 1536, 0);
    // level 4: C=512, N=M=144
    pack_at<<<dim3((512 * 144 + 255) / 256, B), 256, 0, stream>>>(s4s, a4, 512, 144);
    tnorm<<<dim3(1, B), 256, 0, stream>>>(s4t, tn4, 512, 144);
    conv(s4t, a4, nullptr, nullptr, G4, 512, 1, 144, 144, 1, 144,
         1, 144, 0, 0, 1, 144, 0, (long long)144 * 512);
    nn_concat<<<dim3(1, B), 256, 0, stream>>>(s4s, s4t, G4, tn4, c4, 512, 144, 144, 1024, 0);
    // upsample c4 12x12 -> 28x28 directly into dcat channels [512,1536)
    bilinear<<<dim3((1024 * 784 + 255) / 256, B), 256, 0, stream>>>(
        c4, dcat, 1024, 12, 12, 28, 28, 1536, 512);

    // ------------------------------- decoder -------------------------------
    // dec3
    conv(dcat, dw1[0], dsc1[0], dsh1[0], X, 1536, 28, 28, 512, 26, 26,
         3, 26, 0, 0, 1, 676, 1, 0);
    conv(X, dw2[0], dsc2[0], dsh2[0], Y, 512, 26, 26, 512, 24, 24,
         3, 24, 0, 0, 1, 576, 1, 0);
    for (int k = 0; k < 2; ++k)
        for (int l = 0; l < 2; ++l)
            conv(Y, dwt[0] + (size_t)(k * 2 + l) * 256 * 512, nullptr, dcp[0][5], P1,
                 512, 24, 24, 256, 24, 24, 1, 48, k, l, 2, 2304, 0, 0);
    // dec2
    conv(P1, dw1[1], dsc1[1], dsh1[1], X, 256, 48, 48, 128, 46, 46,
         3, 46, 0, 0, 1, 2116, 1, 0);
    conv(X, dw2[1], dsc2[1], dsh2[1], Y, 128, 46, 46, 128, 44, 44,
         3, 44, 0, 0, 1, 1936, 1, 0);
    for (int k = 0; k < 2; ++k)
        for (int l = 0; l < 2; ++l)
            conv(Y, dwt[1] + (size_t)(k * 2 + l) * 64 * 128, nullptr, dcp[1][5], X,
                 128, 44, 44, 64, 44, 44, 1, 88, k, l, 2, 7744, 0, 0);
    // dec1: 1x1 conv (Cout=1) then bilinear 88 -> 256
    conv1x1<<<dim3((7744 + 255) / 256, B), 256, 0, stream>>>(X, dec1w, dec1b, P2, 64, 7744);
    bilinear<<<dim3((256 * 256 + 255) / 256, B), 256, 0, stream>>>(
        P2, (float*)d_out, 1, 88, 88, 256, 256, 1, 0);
}
